// DGATHead_T_48747878809860
// MI455X (gfx1250) — matrix operand order repaired and checked by reading, not compiled, not run on hardware
//
#include <hip/hip_runtime.h>
#include <hip/hip_bf16.h>
#include <math.h>

#define N_NODES 8192
#define D_IN    512
#define D_OUT   64
#define WAVES   8
#define NEG_INF_F (-9.0e15f)

typedef __attribute__((ext_vector_type(16))) __bf16 v16bf;
typedef __attribute__((ext_vector_type(8)))  float  v8f;

// K-slot -> j offset within a 64-wide tile, contiguous k = 16*hf + (slot & 15), same order as B.
// Slots 0..15 -> P0 (j offsets 0..31), slots 16..31 -> P1 (j offsets 32..63).
__device__ __forceinline__ int jloff(int s, int hf) {
    int sl = s & 15;
    int p  = sl >> 1;
    int q  = sl & 1;
    return sl + hf * 16 + (s >> 4) * 32;
}

// ---------------------------------------------------------------------------
// Kernel 1: h = input @ w  (bf16 WMMA, f32 accumulate)
//   - writes hT (bf16, [64][8192]) for the aggregation kernel's B operand
//   - reduces s_src[i] = h[i].a_src, s_dst[i] = h[i].a_dst from f32 accums
// ---------------------------------------------------------------------------
__global__ __launch_bounds__(32)
void gat_h_kernel(const float* __restrict__ inp,
                  const float* __restrict__ w,
                  const float* __restrict__ a,
                  __bf16* __restrict__ hT,
                  float* __restrict__ s_src,
                  float* __restrict__ s_dst) {
    const int lane = threadIdx.x;
    const int hf   = lane >> 4;
    const int l16  = lane & 15;
    const int i0   = blockIdx.x * 16;

    v8f acc[4] = {};

    for (int kb = 0; kb < D_IN; kb += 32) {
        v16bf A;
        const float* arow = inp + (size_t)(i0 + l16) * D_IN + kb + hf * 16;
#pragma unroll
        for (int p = 0; p < 8; ++p) {
            int k = 2 * p;
            A[2 * p]     = (__bf16)arow[k];
            A[2 * p + 1] = (__bf16)arow[k + 1];
        }
#pragma unroll
        for (int t = 0; t < 4; ++t) {
            v16bf B;
            const float* wcol = w + (size_t)(kb + hf * 16) * D_OUT + t * 16 + l16;
#pragma unroll
            for (int v = 0; v < 8; ++v) {
                B[2 * v]     = (__bf16)wcol[(2 * v) * D_OUT];
                B[2 * v + 1] = (__bf16)wcol[(2 * v + 1) * D_OUT];
            }
            acc[t] = __builtin_amdgcn_wmma_f32_16x16x32_bf16(
                false, A, false, B, (short)0, acc[t], false, false);
        }
    }

#pragma unroll
    for (int t = 0; t < 4; ++t) {
        int f = t * 16 + l16;
#pragma unroll
        for (int r = 0; r < 8; ++r)
            hT[(size_t)f * N_NODES + i0 + r + 8 * hf] = (__bf16)acc[t][r];
    }

    float ssrc[8], sdst[8];
#pragma unroll
    for (int r = 0; r < 8; ++r) { ssrc[r] = 0.f; sdst[r] = 0.f; }
#pragma unroll
    for (int t = 0; t < 4; ++t) {
        int f = t * 16 + l16;
        float as = a[f];
        float ad = a[D_OUT + f];
#pragma unroll
        for (int r = 0; r < 8; ++r) {
            ssrc[r] += acc[t][r] * as;
            sdst[r] += acc[t][r] * ad;
        }
    }
#pragma unroll
    for (int msk = 1; msk <= 8; msk <<= 1) {
#pragma unroll
        for (int r = 0; r < 8; ++r) {
            ssrc[r] += __shfl_xor(ssrc[r], msk, 32);
            sdst[r] += __shfl_xor(sdst[r], msk, 32);
        }
    }
    if (l16 == 0) {
#pragma unroll
        for (int r = 0; r < 8; ++r) {
            int row = i0 + r + 8 * hf;
            s_src[row] = ssrc[r];
            s_dst[row] = sdst[r];
        }
    }
}

// ---------------------------------------------------------------------------
// Kernel 2: fused masked-softmax aggregation, flash style, 64-j tiles.
//   - adj tile double-buffered (only HBM-latency loads)
//   - softmax denominator accumulated via extra WMMA against a ones matrix
// ---------------------------------------------------------------------------
__global__ __launch_bounds__(WAVES * 32)
void gat_attn_kernel(const int* __restrict__ adj,
                     const __bf16* __restrict__ hT,
                     const float* __restrict__ s_src,
                     const float* __restrict__ s_dst,
                     float* __restrict__ out) {
    __shared__ float lds_acc[WAVES][16][D_OUT];
    __shared__ float lds_m[WAVES][16];
    __shared__ float lds_l[WAVES][16];
    __shared__ float lds_M[16];
    __shared__ float lds_L[16];

    const int tid  = threadIdx.x;
    const int wave = tid >> 5;
    const int lane = tid & 31;
    const int hf   = lane >> 4;
    const int l16  = lane & 15;
    const int i0   = blockIdx.x * 16;
    const int jt0  = wave * 64;
    const int STRIDE = WAVES * 64;

    const float ss = s_src[i0 + l16];
    float m = -INFINITY;
    v8f acc[4] = {};
    v8f accl = {};          // denominator accumulator (row sums)

    v16bf Bones;
#pragma unroll
    for (int s = 0; s < 16; ++s) Bones[s] = (__bf16)1.0f;

    const int* adjcol = adj + i0 + l16;   // column for this lane's row

    // prologue: load adj for first tile
    int adjv[32];
#pragma unroll
    for (int s = 0; s < 32; ++s)
        adjv[s] = adjcol[(size_t)(jt0 + jloff(s, hf)) * N_NODES];

    for (int jt = jt0; jt < N_NODES; jt += STRIDE) {
        // issue next tile's adj loads (clamped dummy on last iteration)
        int jn = jt + STRIDE;
        int jnc = (jn < N_NODES) ? jn : jt0;
        int adjn[32];
#pragma unroll
        for (int s = 0; s < 32; ++s)
            adjn[s] = adjcol[(size_t)(jnc + jloff(s, hf)) * N_NODES];

        // scores for this lane's row over the 64-j tile, in A-slot order
        float e[32];
        float tmax = -INFINITY;
#pragma unroll
        for (int s = 0; s < 32; ++s) {
            int j = jt + jloff(s, hf);
            float x = ss + s_dst[j];
            x = fmaxf(x, 0.2f * x);                 // leaky_relu (0.2 < 1)
            float ev = (adjv[s] > 0) ? x : NEG_INF_F;
            e[s] = ev;
            tmax = fmaxf(tmax, ev);
        }
        tmax = fmaxf(tmax, __shfl_xor(tmax, 16, 32));
        float mnew  = fmaxf(m, tmax);               // always finite here
        float scale = __expf(m - mnew);
        m = mnew;

        v16bf P0, P1;
#pragma unroll
        for (int s = 0; s < 16; ++s) {
            P0[s] = (__bf16)__expf(e[s] - mnew);
            P1[s] = (__bf16)__expf(e[16 + s] - mnew);
        }

        // rescale accumulators: row r + 8*hf's scale lives in lane (r + 8*hf)
        float rsc[8];
#pragma unroll
        for (int r = 0; r < 8; ++r)
            rsc[r] = __shfl(scale, 8 * hf + r, 32);
#pragma unroll
        for (int r = 0; r < 8; ++r) {
            accl[r] *= rsc[r];
#pragma unroll
            for (int t = 0; t < 4; ++t)
                acc[t][r] *= rsc[r];
        }

        // acc += P0 @ H[jt..jt+31] + P1 @ H[jt+32..jt+63]
#pragma unroll
        for (int t = 0; t < 4; ++t) {
            v16bf B0, B1;
            const __bf16* hrow =
                hT + (size_t)(t * 16 + l16) * N_NODES + jt + hf * 16;
#pragma unroll
            for (int v = 0; v < 8; ++v) {
                B0[2 * v]     = hrow[2 * v];
                B0[2 * v + 1] = hrow[2 * v + 1];
                B1[2 * v]     = hrow[32 + 2 * v];
                B1[2 * v + 1] = hrow[32 + 2 * v + 1];
            }
            acc[t] = __builtin_amdgcn_wmma_f32_16x16x32_bf16(
                false, P0, false, B0, (short)0, acc[t], false, false);
            acc[t] = __builtin_amdgcn_wmma_f32_16x16x32_bf16(
                false, P1, false, B1, (short)0, acc[t], false, false);
        }
        // denominator: accl += P0 @ ones + P1 @ ones  (row sums)
        accl = __builtin_amdgcn_wmma_f32_16x16x32_bf16(
            false, P0, false, Bones, (short)0, accl, false, false);
        accl = __builtin_amdgcn_wmma_f32_16x16x32_bf16(
            false, P1, false, Bones, (short)0, accl, false, false);

#pragma unroll
        for (int s = 0; s < 32; ++s) adjv[s] = adjn[s];
    }

    // Spill per-wave state and combine across waves.
    if (hf == 0) lds_m[wave][l16] = m;
    if (l16 == 0) {
#pragma unroll
        for (int r = 0; r < 8; ++r)
            lds_l[wave][r + 8 * hf] = accl[r];
    }
#pragma unroll
    for (int t = 0; t < 4; ++t)
#pragma unroll
        for (int r = 0; r < 8; ++r)
            lds_acc[wave][r + 8 * hf][t * 16 + l16] = acc[t][r];
    __syncthreads();

    if (tid < 16) {
        float M = -INFINITY;
        for (int wv = 0; wv < WAVES; ++wv) M = fmaxf(M, lds_m[wv][tid]);
        float L = 0.f;
        for (int wv = 0; wv < WAVES; ++wv)
            L += lds_l[wv][tid] * __expf(lds_m[wv][tid] - M);
        lds_M[tid] = M;
        lds_L[tid] = L;
    }
    __syncthreads();

    {
        int row = tid >> 4;
        int f0  = (tid & 15) * 4;
        float M = lds_M[row];
        float invL = 1.0f / lds_L[row];
        float o[4] = {0.f, 0.f, 0.f, 0.f};
        for (int wv = 0; wv < WAVES; ++wv) {
            float sc = __expf(lds_m[wv][row] - M);
#pragma unroll
            for (int q = 0; q < 4; ++q)
                o[q] += lds_acc[wv][row][f0 + q] * sc;
        }
#pragma unroll
        for (int q = 0; q < 4; ++q) {
            float v = o[q] * invL;
            v = (v > 0.f) ? v : (__expf(v) - 1.0f);   // elu
            out[(size_t)(i0 + row) * D_OUT + f0 + q] = v;
        }
    }
}

// ---------------------------------------------------------------------------
extern "C" void kernel_launch(void* const* d_in, const int* in_sizes, int n_in,
                              void* d_out, int out_size, void* d_ws, size_t ws_size,
                              hipStream_t stream) {
    const float* inp = (const float*)d_in[0];   // [8192][512] f32
    const int*   adj = (const int*)  d_in[1];   // [8192][8192] i32
    const float* w   = (const float*)d_in[2];   // [512][64] f32
    const float* a   = (const float*)d_in[3];   // [128][1] f32
    float*       out = (float*)d_out;           // [8192][64] f32

    char* ws = (char*)d_ws;
    __bf16* hT   = (__bf16*)ws;                                   // 1 MB
    float*  ssrc = (float*)(ws + (size_t)D_OUT * N_NODES * 2);    // 32 KB
    float*  sdst = ssrc + N_NODES;                                // 32 KB

    gat_h_kernel<<<N_NODES / 16, 32, 0, stream>>>(inp, w, a, hT, ssrc, sdst);
    gat_attn_kernel<<<N_NODES / 16, WAVES * 32, 0, stream>>>(adj, hT, ssrc, sdst, out);
}
